// UpSample_1821066133670
// MI455X (gfx1250) — matrix operand order repaired and checked by reading, not compiled, not run on hardware
//
#include <hip/hip_runtime.h>
#include <stdint.h>

// ---------------- types ----------------
typedef __attribute__((ext_vector_type(16))) __bf16    v16bf;
typedef __attribute__((ext_vector_type(8)))  float     v8f;
typedef __attribute__((ext_vector_type(4)))  unsigned  u32x4;

union BFrag { v16bf v; unsigned u[8]; u32x4 q[2]; };

#define BS    32
#define NSTK  32
#define NBEF  64
#define COORD 32
#define SPIN  1024
#define SPOUT 512
#define DNIN  256
#define DNOUT 128
#define NSP   64
#define WOUT  128
#define EPSBN 1e-5f

__device__ __forceinline__ unsigned short f2bf(float f) {
  unsigned u = __float_as_uint(f);
  unsigned r = u + 0x7FFFu + ((u >> 16) & 1u);   // round-to-nearest-even
  return (unsigned short)(r >> 16);
}

// ---------------- K0: convert weights to bf16 ----------------
// spw_bf[o*1024+c] = bf16(sp_w[o,c]);  wc_bf[(k*128+co)*256+ci] = bf16(ct_w[ci,co,0,k])
__global__ void k_convw(const float* __restrict__ spw, const float* __restrict__ ctw,
                        unsigned short* __restrict__ spw_bf, unsigned short* __restrict__ wc_bf) {
  int i = blockIdx.x * blockDim.x + threadIdx.x;
  if (i < SPOUT * SPIN) spw_bf[i] = f2bf(spw[i]);
  if (i < 4 * DNOUT * DNIN) {
    int k = i >> 15;            // / (128*256)
    int rem = i & 32767;
    int co = rem >> 8, ci = rem & 255;
    wc_bf[i] = f2bf(ctw[(ci * DNOUT + co) * 4 + k]);
  }
}

// ---------------- K1: top-2 neighbors + inverse-distance weights ----------------
// one wave per (b, n_bef); lane m computes dist to stk_coor[b,m,:]
__global__ void k_topk(const float* __restrict__ coor, const float* __restrict__ bef,
                       int* __restrict__ idx, float* __restrict__ wgt) {
  int gw   = (blockIdx.x * blockDim.x + threadIdx.x) >> 5;
  int lane = threadIdx.x & 31;
  int b = gw >> 6, n = gw & 63;
  const float* bp = bef  + (b * NBEF + n)    * COORD;
  const float* cp = coor + (b * NSTK + lane) * COORD;
  float d = 0.f;
  #pragma unroll 8
  for (int c = 0; c < COORD; ++c) { float t = bp[c] - cp[c]; d += t * t; }

  float v0 = d; int i0 = lane;
  for (int off = 16; off; off >>= 1) {
    float ov = __shfl_xor(v0, off, 32); int oi = __shfl_xor(i0, off, 32);
    if (ov < v0 || (ov == v0 && oi < i0)) { v0 = ov; i0 = oi; }
  }
  float d2 = (lane == i0) ? 3.4e38f : d;
  float v1 = d2; int i1 = lane;
  for (int off = 16; off; off >>= 1) {
    float ov = __shfl_xor(v1, off, 32); int oi = __shfl_xor(i1, off, 32);
    if (ov < v1 || (ov == v1 && oi < i1)) { v1 = ov; i1 = oi; }
  }
  if (lane == 0) {
    float r0 = 1.f / (v0 + 1e-8f), r1 = 1.f / (v1 + 1e-8f);
    float s  = r0 + r1;
    idx[2 * gw] = i0;      idx[2 * gw + 1] = i1;
    wgt[2 * gw] = r0 / s;  wgt[2 * gw + 1] = r1 / s;
  }
}

// ---------------- K2: sparse interpolation -> S_t bf16 [b][n][c] (K-contiguous) ----------------
__global__ void k_sinterp(const float* __restrict__ sf, const int* __restrict__ idx,
                          const float* __restrict__ wgt, unsigned short* __restrict__ st) {
  int bn = blockIdx.x; int b = bn >> 6;
  int i0 = idx[2 * bn], i1 = idx[2 * bn + 1];
  float w0 = wgt[2 * bn], w1 = wgt[2 * bn + 1];
  const float* base = sf + (long)b * SPIN * NSTK;
  unsigned short* out = st + (long)bn * SPIN;
  for (int c = threadIdx.x; c < SPIN; c += blockDim.x) {
    float v = w0 * base[c * NSTK + i0] + w1 * base[c * NSTK + i1];
    out[c] = f2bf(v);
  }
}

// ---------------- K3: sparse GEMM h[b,512,64] = sp_w @ S  (WMMA bf16) ----------------
__global__ void __launch_bounds__(256) k_sgemm(const unsigned short* __restrict__ spw_bf,
                                               const unsigned short* __restrict__ st,
                                               const float* __restrict__ spb,
                                               float* __restrict__ hout) {
  int g    = (blockIdx.x * blockDim.x + threadIdx.x) >> 5;
  int lane = threadIdx.x & 31, hi = lane >> 4, lm = lane & 15;
  int b = g >> 7; int rem = g & 127; int mt = rem >> 2; int nt = rem & 3;

  const unsigned short* abase = spw_bf + (mt * 16 + lm) * SPIN;
  const unsigned short* bbase = st + ((long)(b * NBEF + nt * 16 + lm)) * SPIN + 8 * hi;

  v8f acc = {0.f, 0.f, 0.f, 0.f, 0.f, 0.f, 0.f, 0.f};
  for (int ks = 0; ks < SPIN / 32; ++ks) {
    int k0 = ks * 32;
    BFrag a, bb;
    #pragma unroll
    for (int v = 0; v < 8; ++v) {  // A 16x32 bf16 per-lane layout (ISA 7.12.2)
      int kb = k0 + ((v < 4) ? (2 * v + 8 * hi) : (16 + 2 * (v - 4) + 8 * hi));
      a.u[v] = *(const unsigned*)(abase + kb);
    }
    bb.q[0] = *(const u32x4*)(bbase + k0);
    bb.q[1] = *(const u32x4*)(bbase + k0 + 16);
    acc = __builtin_amdgcn_wmma_f32_16x16x32_bf16(false, a.v, false, bb.v,
                                                  (short)0, acc, false, false);
  }
  int n = nt * 16 + lm;
  #pragma unroll
  for (int r = 0; r < 8; ++r) {
    int co = mt * 16 + r + 8 * hi;
    hout[(b * SPOUT + co) * NBEF + n] = acc[r] + spb[co];
  }
}

// ---------------- K4: BN1d batch stats (per channel over b,n) ----------------
__global__ void k_stats1(const float* __restrict__ h, float* __restrict__ st) {
  int c = blockIdx.x;
  float s = 0.f, sq = 0.f;
  for (int i = threadIdx.x; i < BS * NBEF; i += blockDim.x) {
    int b = i >> 6, n = i & 63;
    float v = h[(b * SPOUT + c) * NBEF + n];
    s += v; sq += v * v;
  }
  __shared__ float ls[256], lq[256];
  ls[threadIdx.x] = s; lq[threadIdx.x] = sq;
  __syncthreads();
  for (int off = 128; off; off >>= 1) {
    if (threadIdx.x < off) { ls[threadIdx.x] += ls[threadIdx.x + off];
                             lq[threadIdx.x] += lq[threadIdx.x + off]; }
    __syncthreads();
  }
  if (threadIdx.x == 0) {
    float mu  = ls[0] * (1.f / (BS * NBEF));
    float var = lq[0] * (1.f / (BS * NBEF)) - mu * mu;
    st[2 * c] = mu; st[2 * c + 1] = var;
  }
}

// ---------------- K5: BN1d + ReLU in place ----------------
__global__ void k_bnrelu1(float* __restrict__ h, const float* __restrict__ st,
                          const float* __restrict__ g, const float* __restrict__ be) {
  int i = blockIdx.x * blockDim.x + threadIdx.x;
  int c = (i >> 6) & (SPOUT - 1);
  float v = g[c] * (h[i] - st[2 * c]) * rsqrtf(st[2 * c + 1] + EPSBN) + be[c];
  h[i] = v > 0.f ? v : 0.f;
}

// ---------------- K6: dense interpolation -> X_t bf16 [b][n][sp][ci] ----------------
__global__ void k_dinterp(const float* __restrict__ df, const int* __restrict__ idx,
                          const float* __restrict__ wgt, unsigned short* __restrict__ xt) {
  __shared__ unsigned short tile[NSP][DNIN + 8];   // padded to dodge bank conflicts
  int bn = blockIdx.x; int b = bn >> 6;
  int i0 = idx[2 * bn], i1 = idx[2 * bn + 1];
  float w0 = wgt[2 * bn], w1 = wgt[2 * bn + 1];
  const float* db = df + (long)b * DNIN * NSTK * NSP;
  for (int e = 0; e < 64; ++e) {                   // coalesced gather over sp
    int lin = e * 256 + threadIdx.x;
    int ci = lin >> 6, sp = lin & 63;
    float v = w0 * db[(ci * NSTK + i0) * NSP + sp] + w1 * db[(ci * NSTK + i1) * NSP + sp];
    tile[sp][ci] = f2bf(v);
  }
  __syncthreads();
  unsigned short* out = xt + (long)bn * NSP * DNIN;
  for (int e = 0; e < 32; ++e) {                   // coalesced transposed store (dword)
    int lin = e * 256 + threadIdx.x;
    int sp = lin >> 7, cp = lin & 127;
    *(unsigned*)(out + sp * DNIN + cp * 2) = *(const unsigned*)(&tile[sp][cp * 2]);
  }
}

// ---------------- K7: conv-transpose as 4-tap WMMA GEMM + shuffle epilogue ----------------
// P_k[co,j] = sum_ci W_k[co,ci]*x[ci,j];  y[2j]=P1[j]+P3[j-1]+b;  y[2j+1]=P2[j]+P0[j+1]+b
__global__ void __launch_bounds__(256) k_cgemm(const unsigned short* __restrict__ wc_bf,
                                               const unsigned short* __restrict__ xt,
                                               const float* __restrict__ ctb,
                                               float* __restrict__ y) {
  __shared__ unsigned short Xs[NSP][DNIN + 8];     // 64x264 bf16 X tile for this (b,n)
  int bn = blockIdx.x; int b = bn >> 6; int n = bn & 63;
  int tid = threadIdx.x;
  const unsigned short* xbase = xt + (long)bn * NSP * DNIN;

  // ---- CDNA5 async staging: memory -> LDS directly (ASYNCcnt), no VGPR round-trip ----
  // GVS mode: LDS byte address from VDST; global address = SGPR base + 32-bit VGPR offset
  {
    unsigned ldsbase = (unsigned)(unsigned long long)(&Xs[0][0]);
    #pragma unroll
    for (int e = 0; e < 8; ++e) {
      int lin = e * 256 + tid;               // 2048 chunks of 16 B (64 rows x 512 B)
      int sp = lin >> 5, cq = lin & 31;      // row, 16B-chunk within row
      unsigned dst = ldsbase + (unsigned)(sp * (DNIN + 8) * 2 + cq * 16);
      unsigned srcoff = (unsigned)lin * 16u; // byte offset from block-uniform base
      asm volatile("global_load_async_to_lds_b128 %0, %1, %2"
                   :: "v"(dst), "v"(srcoff), "s"(xbase) : "memory");
    }
    asm volatile("s_wait_asynccnt 0x0" ::: "memory");
  }
  __syncthreads();

  int w = tid >> 5, lane = tid & 31, hi = lane >> 4, lm = lane & 15;
  int mt = w;                                      // wave = one 16-row co tile
  v8f acc[4][4];                                   // [tap k][n-tile]
  #pragma unroll
  for (int k = 0; k < 4; ++k)
    #pragma unroll
    for (int nt = 0; nt < 4; ++nt) acc[k][nt] = (v8f){0.f,0.f,0.f,0.f,0.f,0.f,0.f,0.f};

  for (int ks = 0; ks < DNIN / 32; ++ks) {
    int k0 = ks * 32;
    BFrag a[4];
    #pragma unroll
    for (int k = 0; k < 4; ++k) {
      const unsigned short* ab = wc_bf + (k * DNOUT + mt * 16 + lm) * DNIN;
      #pragma unroll
      for (int v = 0; v < 8; ++v) {
        int kb = k0 + ((v < 4) ? (2 * v + 8 * hi) : (16 + 2 * (v - 4) + 8 * hi));
        a[k].u[v] = *(const unsigned*)(ab + kb);
      }
    }
    #pragma unroll
    for (int nt = 0; nt < 4; ++nt) {
      BFrag bb;
      const unsigned short* bp = &Xs[nt * 16 + lm][k0 + 8 * hi];
      bb.q[0] = *(const u32x4*)bp;
      bb.q[1] = *(const u32x4*)(bp + 16);
      #pragma unroll
      for (int k = 0; k < 4; ++k)
        acc[k][nt] = __builtin_amdgcn_wmma_f32_16x16x32_bf16(false, a[k].v, false, bb.v,
                                                             (short)0, acc[k][nt], false, false);
    }
  }

  // epilogue: neighbor columns via wave shuffles (j = nt*16 + lm, lane = hi*16 + lm)
  #pragma unroll
  for (int nt = 0; nt < 4; ++nt) {
    int j = nt * 16 + lm;
    #pragma unroll
    for (int r = 0; r < 8; ++r) {
      int cl = r + 8 * hi;
      int co = mt * 16 + cl;
      float p3_same = __shfl_up(acc[3][nt][r], 1, 32);                 // j-1 in same tile
      float p3_prev = __shfl(acc[3][nt > 0 ? nt - 1 : 0][r], hi * 16 + 15, 32);
      float p0_same = __shfl_down(acc[0][nt][r], 1, 32);               // j+1 in same tile
      float p0_next = __shfl(acc[0][nt < 3 ? nt + 1 : 3][r], hi * 16 + 0, 32);
      float p3m = (lm > 0) ? p3_same : ((nt > 0) ? p3_prev : 0.f);
      float p0p = (lm < 15) ? p0_same : ((nt < 3) ? p0_next : 0.f);
      float bias = ctb[co];
      float ye = acc[1][nt][r] + p3m + bias;
      float yo = acc[2][nt][r] + p0p + bias;
      float2 o2 = make_float2(ye, yo);
      *(float2*)(y + ((long)(b * DNOUT + co) * NBEF + n) * WOUT + 2 * j) = o2;
    }
  }
}

// ---------------- K8: BN2d batch stats ----------------
__global__ void k_stats2(const float* __restrict__ y, float* __restrict__ st) {
  int c = blockIdx.x;
  float s = 0.f, sq = 0.f;
  for (int i = threadIdx.x; i < BS * NBEF * WOUT; i += blockDim.x) {
    int b = i >> 13, rem = i & 8191;
    float v = y[((long)(b * DNOUT + c) << 13) + rem];
    s += v; sq += v * v;
  }
  __shared__ float ls[256], lq[256];
  ls[threadIdx.x] = s; lq[threadIdx.x] = sq;
  __syncthreads();
  for (int off = 128; off; off >>= 1) {
    if (threadIdx.x < off) { ls[threadIdx.x] += ls[threadIdx.x + off];
                             lq[threadIdx.x] += lq[threadIdx.x + off]; }
    __syncthreads();
  }
  if (threadIdx.x == 0) {
    const float inv = 1.f / (BS * NBEF * WOUT);
    float mu  = ls[0] * inv;
    float var = lq[0] * inv - mu * mu;
    st[2 * c] = mu; st[2 * c + 1] = var;
  }
}

// ---------------- K9: BN2d + ReLU in place ----------------
__global__ void k_bnrelu2(float* __restrict__ y, const float* __restrict__ st,
                          const float* __restrict__ g, const float* __restrict__ be) {
  long i = (long)blockIdx.x * blockDim.x + threadIdx.x;
  int c = (int)((i >> 13) & (DNOUT - 1));
  float v = g[c] * (y[i] - st[2 * c]) * rsqrtf(st[2 * c + 1] + EPSBN) + be[c];
  y[i] = v > 0.f ? v : 0.f;
}

// ---------------- host launcher ----------------
extern "C" void kernel_launch(void* const* d_in, const int* in_sizes, int n_in,
                              void* d_out, int out_size, void* d_ws, size_t ws_size,
                              hipStream_t stream) {
  const float* sparse_fea = (const float*)d_in[0];
  const float* dense_fea  = (const float*)d_in[1];
  const float* stk_coor   = (const float*)d_in[2];
  const float* stk_bef    = (const float*)d_in[3];
  const float* sp_w       = (const float*)d_in[4];
  const float* sp_b       = (const float*)d_in[5];
  const float* sp_gamma   = (const float*)d_in[6];
  const float* sp_beta    = (const float*)d_in[7];
  const float* ct_w       = (const float*)d_in[8];
  const float* ct_b       = (const float*)d_in[9];
  const float* bn_gamma   = (const float*)d_in[10];
  const float* bn_beta    = (const float*)d_in[11];

  char* ws = (char*)d_ws;
  size_t off = 0;
  int*            w_idx  = (int*)(ws + off);            off += 2048 * 2 * sizeof(int);
  float*          w_wgt  = (float*)(ws + off);          off += 2048 * 2 * sizeof(float);
  unsigned short* spw_bf = (unsigned short*)(ws + off); off += (size_t)SPOUT * SPIN * 2;
  unsigned short* wc_bf  = (unsigned short*)(ws + off); off += (size_t)4 * DNOUT * DNIN * 2;
  unsigned short* s_t    = (unsigned short*)(ws + off); off += (size_t)BS * NBEF * SPIN * 2;
  float*          st1    = (float*)(ws + off);          off += SPOUT * 2 * sizeof(float);
  float*          st2    = (float*)(ws + off);          off += DNOUT * 2 * sizeof(float);
  off = (off + 255) & ~(size_t)255;
  unsigned short* x_t    = (unsigned short*)(ws + off); // 64 MiB: [32][64][64][256] bf16

  float* h = (float*)d_out;                       // [32,512,64] pre-BN, normalized in place
  float* y = (float*)d_out + (size_t)BS * SPOUT * NBEF;  // [32,128,64,128]

  k_convw  <<<2048,   256, 0, stream>>>(sp_w, ct_w, spw_bf, wc_bf);
  k_topk   <<<256,    256, 0, stream>>>(stk_coor, stk_bef, w_idx, w_wgt);
  k_sinterp<<<2048,   256, 0, stream>>>(sparse_fea, w_idx, w_wgt, s_t);
  k_sgemm  <<<512,    256, 0, stream>>>(spw_bf, s_t, sp_b, h);
  k_stats1 <<<512,    256, 0, stream>>>(h, st1);
  k_bnrelu1<<<4096,   256, 0, stream>>>(h, st1, sp_gamma, sp_beta);
  k_dinterp<<<2048,   256, 0, stream>>>(dense_fea, w_idx, w_wgt, x_t);
  k_cgemm  <<<2048,   256, 0, stream>>>(wc_bf, x_t, ct_b, y);
  k_stats2 <<<128,    256, 0, stream>>>(y, st2);
  k_bnrelu2<<<131072, 256, 0, stream>>>(y, st2, bn_gamma, bn_beta);
}
